// router_54365696033066
// MI455X (gfx1250) — matrix operand order repaired and checked by reading, not compiled, not run on hardware
//
#include <hip/hip_runtime.h>
#include <hip/hip_bf16.h>
#include <math.h>

// ---------------------------------------------------------------------------
// MoE router for MI455X (gfx1250, wave32, WMMA).
//   probs = scatter(softmax(top2(gelu(x@w1+b1)@w2 + (te[task]@w2 + b2))))
// Memory-bound: 134MB x-read dominates (~6us @ 23.3TB/s). Compute in bf16
// WMMA 16x16x32. x is register-pipelined (prefetch next chunk during WMMA);
// w1 chunks are staged with async global->LDS copies (ASYNCcnt), double-
// buffered so the copy overlaps compute. In-loop barriers are hand-rolled
// (s_wait_dscnt only) so the backend's conservative "drain all counters
// before s_barrier" does not serialize the x prefetch against the barrier.
// ---------------------------------------------------------------------------

typedef __attribute__((ext_vector_type(16))) __bf16 v16bf;
typedef __attribute__((ext_vector_type(8)))  float  v8f;

#define RDIM   1024   // input dim (K of GEMM1)
#define REMB   128    // hidden dim (N of GEMM1, K of GEMM2)
#define RNC    64     // experts  (N of GEMM2)
#define MTILE  128    // tokens per block
#define KCHUNK 128    // K staged per LDS chunk
#define NKC    (RDIM / KCHUNK)   // 8 chunks
#define XSTR   136    // padded LDS row stride in bf16 elems (272B: bank-spread)

// ---- prep 1: w1t[n][k] = bf16(w1[k][n])  (256KB, L2-resident, shared by all blocks)
__global__ void prep_w1_kernel(const float* __restrict__ w1, __bf16* __restrict__ w1t) {
    int idx = blockIdx.x * blockDim.x + threadIdx.x;  // 0..131071
    int n = idx >> 10;
    int k = idx & 1023;
    w1t[idx] = (__bf16)w1[k * REMB + n];
}

// ---- prep 2: w2t[c][e] = bf16(w2[e][c]);  b2p[c] = b2[c] + sum_e te[task][e]*w2[e][c]
__global__ void prep_w2_kernel(const float* __restrict__ w2, const float* __restrict__ b2,
                               const float* __restrict__ task_emb, const int* __restrict__ task_id,
                               __bf16* __restrict__ w2t, float* __restrict__ b2p) {
    int idx = blockIdx.x * blockDim.x + threadIdx.x;  // 0..8191
    if (idx < REMB * RNC) {
        int c = idx >> 7;
        int e = idx & 127;
        w2t[idx] = (__bf16)w2[e * RNC + c];
    }
    if (idx < RNC) {
        const float* te = task_emb + task_id[0] * REMB;
        float s = b2[idx];
        for (int e = 0; e < REMB; ++e) s += te[e] * w2[e * RNC + idx];
        b2p[idx] = s;
    }
}

// Workgroup barrier that only drains DScnt (LDS visibility). Global loads in
// flight (x register prefetch) are intentionally NOT waited here — they are
// waited by the compiler at their point of use, a full WMMA phase later.
__device__ __forceinline__ void wg_barrier_lds() {
    asm volatile(
        "s_wait_dscnt 0x0\n\t"
        "s_barrier_signal -1\n\t"
        "s_barrier_wait -1" ::: "memory");
}

// Issue 8 async b128 copies: 128B from w1t[gvo ..] into LDS at lds_w.
// Per ISA 10.x: INST_OFFSET is added to BOTH the LDS and the global address,
// so one LDS-address VGPR + one global-offset VGPR + immediate offsets suffice.
__device__ __forceinline__ void async_copy_w1_row(unsigned lds_w, unsigned gvo,
                                                  const __bf16* base) {
    asm volatile(
        "global_load_async_to_lds_b128 %0, %1, %2 offset:0\n\t"
        "global_load_async_to_lds_b128 %0, %1, %2 offset:16\n\t"
        "global_load_async_to_lds_b128 %0, %1, %2 offset:32\n\t"
        "global_load_async_to_lds_b128 %0, %1, %2 offset:48\n\t"
        "global_load_async_to_lds_b128 %0, %1, %2 offset:64\n\t"
        "global_load_async_to_lds_b128 %0, %1, %2 offset:80\n\t"
        "global_load_async_to_lds_b128 %0, %1, %2 offset:96\n\t"
        "global_load_async_to_lds_b128 %0, %1, %2 offset:112"
        :: "v"(lds_w), "v"(gvo), "s"(base) : "memory");
}

// ---- main fused kernel: 256 threads = 8 waves; each wave owns 16 token rows.
__global__ void __launch_bounds__(256)
router_main_kernel(const float* __restrict__ x, const float* __restrict__ b1,
                   const __bf16* __restrict__ w1t, const __bf16* __restrict__ w2t,
                   const float* __restrict__ b2p, float* __restrict__ out) {
    __shared__ __align__(16) __bf16 xbuf[MTILE * XSTR];      // x chunk (bf16), later h
    __shared__ __align__(16) __bf16 wbuf[2][REMB * XSTR];    // w1 chunk [n][k], dbl-buffered
    __shared__ __align__(16) __bf16 w2buf[RNC * XSTR];       // w2t [c][e]
    __shared__ __align__(16) float  lbuf[8 * 16 * 68];       // per-wave logits 16x64 (+pad)

    const int tid  = threadIdx.x;
    const int lane = tid & 31;
    const int wave = tid >> 5;
    const int n    = lane & 15;   // column within 16-wide tile
    const int half = lane >> 4;   // lane half-select (ISA fragment layouts)
    const long t0  = (long)blockIdx.x * MTILE;

    const int row2  = tid >> 1;        // staging: row per thread-pair
    const int cbase = (tid & 1) * 64;  // staging: 64-elem half of the chunk

    // ---- async w1 staging state (per thread: one 128B contiguous run) ------
    const unsigned w1_gvo_base = (unsigned)(row2 * RDIM + cbase) * 2u;  // bytes
    unsigned lds_w[2];
    lds_w[0] = (unsigned)(size_t)(&wbuf[0][row2 * XSTR + cbase]);
    lds_w[1] = (unsigned)(size_t)(&wbuf[1][row2 * XSTR + cbase]);

    // issue async copy of w1 chunk 0 into wbuf[0]
    async_copy_w1_row(lds_w[0], w1_gvo_base, w1t);

    // ---- register-prefetch x chunk 0 ---------------------------------------
    const float* xrow = x + (t0 + row2) * RDIM + cbase;
    float4 xreg[16];
#pragma unroll
    for (int i = 0; i < 16; ++i) xreg[i] = ((const float4*)xrow)[i];

    // stage w2t (64x128 bf16 = 16KB) once: 256 threads x 32 elems
    {
        int c  = tid >> 2;
        int e0 = (tid & 3) * 32;
        const uint4* src = (const uint4*)(w2t + c * REMB + e0);
        uint4*       dst = (uint4*)(w2buf + c * XSTR + e0);
        dst[0] = src[0]; dst[1] = src[1]; dst[2] = src[2]; dst[3] = src[3];
    }

    // per-lane biases for the epilogue columns this lane owns
    float b1v[8];
#pragma unroll
    for (int nt = 0; nt < 8; ++nt) b1v[nt] = b1[nt * 16 + n];
    float b2v[4];
#pragma unroll
    for (int ct = 0; ct < 4; ++ct) b2v[ct] = b2p[ct * 16 + n];

    v8f acc[8];
#pragma unroll
    for (int i = 0; i < 8; ++i) acc[i] = (v8f)0.0f;

    // ================= GEMM1: h = x @ w1, K streamed in 128-chunks ==========
    for (int kc = 0; kc < NKC; ++kc) {
        wg_barrier_lds();   // previous chunk fully consumed; wbuf[(kc+1)&1] free
        // store prefetched x registers -> LDS with f32->bf16 conversion
        {
            __bf16* xd = xbuf + row2 * XSTR + cbase;
#pragma unroll
            for (int i = 0; i < 16; ++i) {
                float4 f = xreg[i];
                union { __bf16 h[4]; uint2 u; } p;
                p.h[0] = (__bf16)f.x; p.h[1] = (__bf16)f.y;
                p.h[2] = (__bf16)f.z; p.h[3] = (__bf16)f.w;
                *(uint2*)(xd + i * 4) = p.u;
            }
        }
        if (kc + 1 < NKC) {
            // prefetch next x chunk into registers (overlaps WMMA below; the
            // in-loop barriers do not drain LOADcnt, so this stays in flight)
            const float* xs = xrow + (kc + 1) * KCHUNK;
#pragma unroll
            for (int i = 0; i < 16; ++i) xreg[i] = ((const float4*)xs)[i];
            // issue async copy of next w1 chunk (overlaps WMMA below)
            async_copy_w1_row(lds_w[(kc + 1) & 1],
                              w1_gvo_base + (unsigned)(kc + 1) * (KCHUNK * 2u), w1t);
            // asyncs complete in order: <=8 outstanding => this chunk's 8 landed
            asm volatile("s_wait_asynccnt 0x8" ::: "memory");
        } else {
            asm volatile("s_wait_asynccnt 0x0" ::: "memory");
        }
        wg_barrier_lds();

        const __bf16* warr = wbuf[kc & 1];
        const __bf16* arow = xbuf + (wave * 16 + n) * XSTR;
#pragma unroll
        for (int ks = 0; ks < 4; ++ks) {
            // A fragment (16x32 bf16): lane holds row M=lane%16,
            // runs K = ks*32 + half*8 + [0,8) and K = ks*32 + 16 + half*8 + [0,8)
            v16bf a;
            *((uint4*)&a)     = *(const uint4*)(arow + ks * 32 + half * 8);
            *((uint4*)&a + 1) = *(const uint4*)(arow + ks * 32 + 16 + half * 8);
#pragma unroll
            for (int nt = 0; nt < 8; ++nt) {
                // B fragment (32x16 bf16): lane holds col N=lane%16,
                // runs K = ks*32 + half*8 + [0,8) and K = ks*32 + 16 + half*8 + [0,8)
                const __bf16* brow = warr + (nt * 16 + n) * XSTR + ks * 32 + half * 8;
                v16bf b;
                *((uint4*)&b)     = *(const uint4*)(brow);
                *((uint4*)&b + 1) = *(const uint4*)(brow + 16);
                acc[nt] = __builtin_amdgcn_wmma_f32_16x16x32_bf16(
                    false, a, false, b, (short)0, acc[nt], false, false);
            }
        }
    }

    // ======== epilogue 1: h = gelu(acc + b1); write bf16 h to LDS ===========
    // C/D layout: lane holds (M = v + 8*half, N = nt*16 + n). Wave-private rows
    // of xbuf are reused; same-wave DS ordering keeps this safe without barrier.
#pragma unroll
    for (int nt = 0; nt < 8; ++nt) {
#pragma unroll
        for (int v = 0; v < 8; ++v) {
            float hv = acc[nt][v] + b1v[nt];
            hv = 0.5f * hv * (1.0f + erff(hv * 0.70710678118654752f));  // exact gelu
            xbuf[(wave * 16 + half * 8 + v) * XSTR + nt * 16 + n] = (__bf16)hv;
        }
    }

    // ================= GEMM2: logits = h @ w2  (K=128) ======================
    v8f acc2[4];
#pragma unroll
    for (int i = 0; i < 4; ++i) acc2[i] = (v8f)0.0f;

    const __bf16* hrow = xbuf + (wave * 16 + n) * XSTR;
#pragma unroll
    for (int ks = 0; ks < 4; ++ks) {
        v16bf a;
        *((uint4*)&a)     = *(const uint4*)(hrow + ks * 32 + half * 8);
        *((uint4*)&a + 1) = *(const uint4*)(hrow + ks * 32 + 16 + half * 8);
#pragma unroll
        for (int ct = 0; ct < 4; ++ct) {
            const __bf16* brow = w2buf + (ct * 16 + n) * XSTR + ks * 32 + half * 8;
            v16bf b;
            *((uint4*)&b)     = *(const uint4*)(brow);
            *((uint4*)&b + 1) = *(const uint4*)(brow + 16);
            acc2[ct] = __builtin_amdgcn_wmma_f32_16x16x32_bf16(
                false, a, false, b, (short)0, acc2[ct], false, false);
        }
    }

    // ======== epilogue 2: +b2', top-2, softmax, scatter =====================
    float* lrow = lbuf + wave * (16 * 68);
#pragma unroll
    for (int ct = 0; ct < 4; ++ct)
#pragma unroll
        for (int v = 0; v < 8; ++v)
            lrow[(half * 8 + v) * 68 + ct * 16 + n] = acc2[ct][v] + b2v[ct];
    wg_barrier_lds();

    if (lane < 16) {
        const float* lr = lrow + lane * 68;
        float v1 = -3.4e38f, v2 = -3.4e38f;
        int   i1 = 0, i2 = 0;
        for (int c = 0; c < RNC; ++c) {
            float vv = lr[c];
            if (vv > v1)      { v2 = v1; i2 = i1; v1 = vv; i1 = c; }  // ties: keep lowest idx
            else if (vv > v2) { v2 = vv; i2 = c; }
        }
        float e2 = expf(v2 - v1);           // <= 1, stable
        float p1 = 1.0f / (1.0f + e2);
        float p2 = e2 * p1;
        float* op = out + (t0 + wave * 16 + lane) * RNC;
#pragma unroll
        for (int c4 = 0; c4 < RNC; c4 += 4) {
            float4 o;
            o.x = (c4 + 0 == i1) ? p1 : ((c4 + 0 == i2) ? p2 : 0.0f);
            o.y = (c4 + 1 == i1) ? p1 : ((c4 + 1 == i2) ? p2 : 0.0f);
            o.z = (c4 + 2 == i1) ? p1 : ((c4 + 2 == i2) ? p2 : 0.0f);
            o.w = (c4 + 3 == i1) ? p1 : ((c4 + 3 == i2) ? p2 : 0.0f);
            ((float4*)op)[c4 >> 2] = o;
        }
    }
}

extern "C" void kernel_launch(void* const* d_in, const int* in_sizes, int n_in,
                              void* d_out, int out_size, void* d_ws, size_t ws_size,
                              hipStream_t stream) {
    const float* x        = (const float*)d_in[0];
    const float* w1       = (const float*)d_in[1];
    const float* b1       = (const float*)d_in[2];
    const float* w2       = (const float*)d_in[3];
    const float* b2       = (const float*)d_in[4];
    const float* task_emb = (const float*)d_in[5];
    const int*   task_id  = (const int*)d_in[6];
    float*       out      = (float*)d_out;

    char*   ws  = (char*)d_ws;
    __bf16* w1t = (__bf16*)ws;                              // 256 KB
    __bf16* w2t = (__bf16*)(ws + 256 * 1024);               // 16 KB
    float*  b2p = (float*)(ws + 256 * 1024 + 16 * 1024);    // 256 B

    const int ntok = in_sizes[0] / RDIM;                    // 32768

    prep_w1_kernel<<<(REMB * RDIM) / 256, 256, 0, stream>>>(w1, w1t);
    prep_w2_kernel<<<(REMB * RNC) / 256, 256, 0, stream>>>(w2, b2, task_emb, task_id, w2t, b2p);
    router_main_kernel<<<ntok / MTILE, 256, 0, stream>>>(x, b1, w1t, w2t, b2p, out);
}
